// PeriodicSAModule3_609885356791
// MI455X (gfx1250) — matrix operand order repaired and checked by reading, not compiled, not run on hardware
//
#include <hip/hip_runtime.h>
#include <hip/hip_bf16.h>

typedef __attribute__((ext_vector_type(16))) __bf16 v16bf;
typedef __attribute__((ext_vector_type(8)))  __bf16 v8bf;
typedef __attribute__((ext_vector_type(8)))  float  v8f;

#define NB      16
#define NATOMS  512
#define CFEAT   64
#define KNN     64
#define NCAND   (NATOMS * 27)     // 13824
#define R2      0.09f
#define H1D     64
#define H2D     64
#define HOUTD   128
#define HP      96                // feature K-dim padded 67 -> 96 (3 x k=32 steps)
#define MAXC    512               // candidate list cap

// ---- workspace layout (bytes) ----
#define OFF_PF   0
#define SZ_PF    (NB * NCAND * 3 * 4)        // 2,654,208 (16B aligned)
#define OFF_W1T  (OFF_PF + SZ_PF)
#define SZ_W1T   (H1D * HP * 2)              // 12,288
#define OFF_W2T  (OFF_W1T + SZ_W1T)
#define SZ_W2T   (H2D * H1D * 2)             // 8,192
#define OFF_W3T  (OFF_W2T + SZ_W2T)
#define SZ_W3T   (HOUTD * H2D * 2)           // 16,384

// ---------------- prep 1: transformed periodic images pf[b][j][3] ----------------
__global__ void prep_pf_kernel(const float* __restrict__ frac,
                               const float* __restrict__ trans,
                               const float* __restrict__ scale,
                               float* __restrict__ pf) {
    int gid = blockIdx.x * blockDim.x + threadIdx.x;
    if (gid >= NB * NCAND) return;
    int b = gid / NCAND;
    int j = gid - b * NCAND;
    int n = j / 27;
    int o = j - n * 27;
    float ox = (float)(o / 9) - 1.0f;          // meshgrid 'ij': o = i*9 + j*3 + k
    float oy = (float)((o / 3) % 3) - 1.0f;
    float oz = (float)(o % 3) - 1.0f;
    const float* f = frac + (b * NATOMS + n) * 3;
    float fx = f[0] + ox, fy = f[1] + oy, fz = f[2] + oz;
    const float* M = trans + b * 9;            // row-major 3x3
    float s = scale[b];
    float px = s * (fx * M[0] + fy * M[3] + fz * M[6]);
    float py = s * (fx * M[1] + fy * M[4] + fz * M[7]);
    float pz = s * (fx * M[2] + fy * M[5] + fz * M[8]);
    float* dst = pf + (size_t)gid * 3;
    dst[0] = px; dst[1] = py; dst[2] = pz;
}

// ---------------- prep 2: weights -> bf16, transposed Wt[n][k] ----------------
__global__ void prep_w_kernel(const float* __restrict__ W1,
                              const float* __restrict__ W2,
                              const float* __restrict__ W3,
                              __bf16* __restrict__ W1t,
                              __bf16* __restrict__ W2t,
                              __bf16* __restrict__ W3t) {
    int gid = blockIdx.x * blockDim.x + threadIdx.x;
    const int N1 = H1D * HP;
    const int N2 = H2D * H1D;
    const int N3 = HOUTD * H2D;
    if (gid < N1) {
        int n = gid / HP, k = gid - n * HP;    // W1: (67, 64) row-major; pad k>=67 with 0
        W1t[gid] = (__bf16)((k < CFEAT + 3) ? W1[k * H1D + n] : 0.0f);
    } else if (gid < N1 + N2) {
        int i = gid - N1;
        int n = i / H1D, k = i - n * H1D;      // W2: (64, 64)
        W2t[i] = (__bf16)W2[k * H2D + n];
    } else if (gid < N1 + N2 + N3) {
        int i = gid - N1 - N2;
        int n = i / H2D, k = i - n * H2D;      // W3: (64, 128)
        W3t[i] = (__bf16)W3[k * HOUTD + n];
    }
}

// ---------------- fragment helpers (wave32 WMMA layouts, ISA 7.12.2) ----------------
__device__ __forceinline__ v8f vzero8() {
    v8f z;
#pragma unroll
    for (int i = 0; i < 8; ++i) z[i] = 0.0f;
    return z;
}

// A tile 16x32 bf16: lanes 0-15 = M rows (K 0..7 then 16..23), lanes 16-31 same M (K 8..15, 24..31)
__device__ __forceinline__ v16bf load_A(const __bf16* __restrict__ base, int stride,
                                        int rowTile, int kb, int mrow, int kh) {
    const __bf16* p = base + (rowTile + mrow) * stride + kb + kh * 8;
    v8bf lo = *(const v8bf*)(p);
    v8bf hi = *(const v8bf*)(p + 16);
    v16bf a;
#pragma unroll
    for (int e = 0; e < 8; ++e) { a[e] = lo[e]; a[e + 8] = hi[e]; }
    return a;
}

// B tile 32x16 bf16 from transposed weights Wt[n][k]: lanes 0-15 K 0..7 then 16..23, lanes 16-31 K 8..15, 24..31
__device__ __forceinline__ v16bf load_B(const __bf16* __restrict__ Wt, int kstride,
                                        int colTile, int kb, int ncol, int kh) {
    const __bf16* p = Wt + (colTile + ncol) * kstride + kb + kh * 8;
    v8bf lo = *(const v8bf*)(p);
    v8bf hi = *(const v8bf*)(p + 16);
    v16bf bv;
#pragma unroll
    for (int e = 0; e < 8; ++e) { bv[e] = lo[e]; bv[e + 8] = hi[e]; }
    return bv;
}

// ---------------- main fused kernel: one workgroup (4 waves) per query atom ----------------
__global__ __launch_bounds__(128) void sa_fused_kernel(
    const float* __restrict__ xg, const float* __restrict__ pf,
    const __bf16* __restrict__ W1t, const __bf16* __restrict__ W2t,
    const __bf16* __restrict__ W3t,
    const float* __restrict__ b1, const float* __restrict__ b2,
    const float* __restrict__ b3,
    float* __restrict__ out)
{
    __shared__ __align__(16) __bf16 sH[64 * HP];      // layer input  (neighbors x 96)
    __shared__ __align__(16) __bf16 sH2[64 * H1D];    // layer-1 activations
    __shared__ float sd2[MAXC];
    __shared__ int   sidx[MAXC];
    __shared__ int   ssel[KNN];
    __shared__ float spmax[4][HOUTD];
    __shared__ int   scount;
    __shared__ int   snv;
    __shared__ float sq[3];

    const int tid  = threadIdx.x;
    const int lane = tid & 31;
    const int w    = tid >> 5;
    const int blk  = blockIdx.x;
    const int b    = blk >> 9;
    const int atom = blk & (NATOMS - 1);
    const float* pfb = pf + (size_t)b * NCAND * 3;

    if (tid == 0) {
        scount = 0;
        const float* qp = pfb + (atom * 27 + 13) * 3;  // center image (0,0,0)
        sq[0] = qp[0]; sq[1] = qp[1]; sq[2] = qp[2];
    }
    if (tid < KNN) ssel[tid] = -1;
    __syncthreads();

    const float qx = sq[0], qy = sq[1], qz = sq[2];

    // ---- radius scan over 13824 candidates ----
    for (int j = tid; j < NCAND; j += 128) {
        float dx = pfb[j * 3 + 0] - qx;
        float dy = pfb[j * 3 + 1] - qy;
        float dz = pfb[j * 3 + 2] - qz;
        float d2 = dx * dx + dy * dy + dz * dz;
        if (d2 <= R2) {
            int s = atomicAdd(&scount, 1);
            if (s < MAXC) { sd2[s] = d2; sidx[s] = j; }
        }
    }
    __syncthreads();
    const int L = min(scount, MAXC);
    if (tid == 0) snv = min(L, KNN);

    // ---- canonical rank-based top-K (tie-break: lower candidate index, as lax.top_k) ----
    for (int i = tid; i < L; i += 128) {
        float di = sd2[i]; int xi = sidx[i];
        int rank = 0;
        for (int jj = 0; jj < L; ++jj) {
            float dj = sd2[jj]; int xj = sidx[jj];
            rank += (dj < di) || ((dj == di) && (xj < xi));
        }
        if (rank < KNN) ssel[rank] = xi;
    }
    __syncthreads();
    const int nv = snv;

    // ---- gather neighbor features into sH (64 x 96 bf16): [x_j (64) | rel (3) | 0-pad] ----
    {
        int r = tid >> 1, half = tid & 1;
        int c0 = half * 48;
        int src = (r < nv) ? ssel[r] : -1;
        const float* xrow = nullptr;
        float rx = 0.f, ry = 0.f, rz = 0.f;
        if (src >= 0) {
            int node = src / 27;
            xrow = xg + ((size_t)b * NATOMS + node) * CFEAT;
            rx = pfb[src * 3 + 0] - qx;
            ry = pfb[src * 3 + 1] - qy;
            rz = pfb[src * 3 + 2] - qz;
        }
        for (int c = c0; c < c0 + 48; ++c) {
            float v = 0.0f;
            if (src >= 0) {
                if (c < CFEAT)          v = xrow[c];
                else if (c == CFEAT)     v = rx;
                else if (c == CFEAT + 1) v = ry;
                else if (c == CFEAT + 2) v = rz;
            }
            sH[r * HP + c] = (__bf16)v;
        }
    }
    __syncthreads();

    const int mrow = lane & 15;   // A: M row / B,C: N column
    const int kh   = lane >> 4;   // K-half selector

    // ---- Layer 1: (64x96) @ (96x64) -> relu -> sH2 ----
    {
        v8f acc[4];
#pragma unroll
        for (int nt = 0; nt < 4; ++nt) acc[nt] = vzero8();
#pragma unroll
        for (int kb = 0; kb < HP; kb += 32) {
            v16bf a = load_A(sH, HP, w * 16, kb, mrow, kh);
#pragma unroll
            for (int nt = 0; nt < 4; ++nt) {
                v16bf bv = load_B(W1t, HP, nt * 16, kb, mrow, kh);
                acc[nt] = __builtin_amdgcn_wmma_f32_16x16x32_bf16(
                    false, a, false, bv, (short)0, acc[nt], false, false);
            }
        }
#pragma unroll
        for (int nt = 0; nt < 4; ++nt) {
            int col = nt * 16 + mrow;
            float bias = b1[col];
#pragma unroll
            for (int r8 = 0; r8 < 8; ++r8) {
                int row = w * 16 + r8 + 8 * kh;
                float v = acc[nt][r8] + bias;
                sH2[row * H1D + col] = (__bf16)(v > 0.f ? v : 0.f);
            }
        }
    }
    __syncthreads();

    // ---- Layer 2: (64x64) @ (64x64) -> relu -> sH (cols 0..63, stride HP) ----
    {
        v8f acc[4];
#pragma unroll
        for (int nt = 0; nt < 4; ++nt) acc[nt] = vzero8();
#pragma unroll
        for (int kb = 0; kb < H1D; kb += 32) {
            v16bf a = load_A(sH2, H1D, w * 16, kb, mrow, kh);
#pragma unroll
            for (int nt = 0; nt < 4; ++nt) {
                v16bf bv = load_B(W2t, H1D, nt * 16, kb, mrow, kh);
                acc[nt] = __builtin_amdgcn_wmma_f32_16x16x32_bf16(
                    false, a, false, bv, (short)0, acc[nt], false, false);
            }
        }
#pragma unroll
        for (int nt = 0; nt < 4; ++nt) {
            int col = nt * 16 + mrow;
            float bias = b2[col];
#pragma unroll
            for (int r8 = 0; r8 < 8; ++r8) {
                int row = w * 16 + r8 + 8 * kh;
                float v = acc[nt][r8] + bias;
                sH[row * HP + col] = (__bf16)(v > 0.f ? v : 0.f);
            }
        }
    }
    __syncthreads();

    // ---- Layer 3: (64x64) @ (64x128), masked row-max ----
    {
        v8f acc[8];
#pragma unroll
        for (int nt = 0; nt < 8; ++nt) acc[nt] = vzero8();
#pragma unroll
        for (int kb = 0; kb < H2D; kb += 32) {
            v16bf a = load_A(sH, HP, w * 16, kb, mrow, kh);
#pragma unroll
            for (int nt = 0; nt < 8; ++nt) {
                v16bf bv = load_B(W3t, H2D, nt * 16, kb, mrow, kh);
                acc[nt] = __builtin_amdgcn_wmma_f32_16x16x32_bf16(
                    false, a, false, bv, (short)0, acc[nt], false, false);
            }
        }
        // masked max over this wave's 16 neighbor rows; C layout: row = r8 + 8*kh, col = mrow
#pragma unroll
        for (int nt = 0; nt < 8; ++nt) {
            float m = -INFINITY;
#pragma unroll
            for (int r8 = 0; r8 < 8; ++r8) {
                int row = w * 16 + r8 + 8 * kh;
                if (row < nv) m = fmaxf(m, acc[nt][r8]);
            }
            float o = __shfl_xor(m, 16);        // combine the two row-halves (same column)
            m = fmaxf(m, o);
            if (lane < 16) spmax[w][nt * 16 + lane] = m;
        }
    }
    __syncthreads();

    // ---- cross-wave max, bias, store ----
    {
        int col = tid;  // 128 columns
        float m = fmaxf(fmaxf(spmax[0][col], spmax[1][col]),
                        fmaxf(spmax[2][col], spmax[3][col]));
        float res = (nv > 0) ? (m + b3[col]) : -INFINITY;
        out[(size_t)blk * HOUTD + col] = res;
    }
}

extern "C" void kernel_launch(void* const* d_in, const int* in_sizes, int n_in,
                              void* d_out, int out_size, void* d_ws, size_t ws_size,
                              hipStream_t stream) {
    const float* x     = (const float*)d_in[0];
    // d_in[1] pos, d_in[2] fps_pos, d_in[3] batch: unused by the reference math
    const float* frac  = (const float*)d_in[4];
    const float* trans = (const float*)d_in[5];
    const float* scale = (const float*)d_in[6];
    const float* W1 = (const float*)d_in[7];
    const float* b1 = (const float*)d_in[8];
    const float* W2 = (const float*)d_in[9];
    const float* b2 = (const float*)d_in[10];
    const float* W3 = (const float*)d_in[11];
    const float* b3 = (const float*)d_in[12];

    char* ws = (char*)d_ws;
    float*  pf  = (float*)(ws + OFF_PF);
    __bf16* W1t = (__bf16*)(ws + OFF_W1T);
    __bf16* W2t = (__bf16*)(ws + OFF_W2T);
    __bf16* W3t = (__bf16*)(ws + OFF_W3T);

    prep_pf_kernel<<<(NB * NCAND + 255) / 256, 256, 0, stream>>>(frac, trans, scale, pf);
    prep_w_kernel<<<(H1D * HP + H2D * H1D + HOUTD * H2D + 255) / 256, 256, 0, stream>>>(
        W1, W2, W3, W1t, W2t, W3t);
    sa_fused_kernel<<<NB * NATOMS, 128, 0, stream>>>(
        x, pf, W1t, W2t, W3t, b1, b2, b3, (float*)d_out);
}
